// PeepholeLSTM_18691697672922
// MI455X (gfx1250) — matrix operand order repaired and checked by reading, not compiled, not run on hardware
//
#include <hip/hip_runtime.h>
#include <hip/hip_bf16.h>
#include <stdint.h>

// ---------------------------------------------------------------------------
// Peephole LSTM (2 layers, T=128, B=256, IN=256, H=512, NC=8) on CDNA5 WMMA.
//
//  * x / weights converted once to f16 (weights stay [N,K] row-major ->
//    B fragments are 32 B contiguous per lane).
//  * Per timestep, per layer, one fused kernel: gate GEMM
//    gates[256,2048] = [x_t | h_prev] @ [Wx | Wh]^T via v_wmma_f32_16x16x32_f16
//    (K split into two regions), then the peephole cell applied directly on
//    the WMMA accumulators, writing c (f32) and h (f16).
//  * Wave = 32 rows x 16 hidden cols x 4 gates = 8 accumulator tiles; each
//    B fragment feeds two WMMAs (12 b128 loads -> 8 WMMAs per K-step).
//  * Workgroup = 8 waves = 32 rows x 128 hidden cols. Grid = (8, 4).
//  * Unconditional global_prefetch_b8 on weight columns one line ahead
//    (keeps the loop branch-free so WMMAs pipeline against loads).
// ---------------------------------------------------------------------------

#define Bsz 256
#define Tsz 128
#define INsz 256
#define Hsz 512

typedef _Float16 v8h  __attribute__((ext_vector_type(8)));
typedef _Float16 v16h __attribute__((ext_vector_type(16)));
typedef float    v8f  __attribute__((ext_vector_type(8)));

__device__ __forceinline__ float sigmoid_f(float x) {
    return 1.0f / (1.0f + __expf(-x));
}
__device__ __forceinline__ float tanh_f(float x) {
    x = fminf(15.0f, fmaxf(-15.0f, x));
    float e = __expf(2.0f * x);
    return (e - 1.0f) / (e + 1.0f);
}

__device__ __forceinline__ v16h load_afrag(const _Float16* p) {
    // halves 0..15 contiguous at p (K order matches contiguous B fragment)
    v8h lo = *(const v8h*)(p);
    v8h hi = *(const v8h*)(p + 8);
    v16h a;
#pragma unroll
    for (int j = 0; j < 8; ++j) { a[j] = lo[j]; a[8 + j] = hi[j]; }
    return a;
}

// ---------------------------------------------------------------------------
// f32 -> f16 elementwise convert (4 elems/thread; n multiple of 4)
// ---------------------------------------------------------------------------
__global__ __launch_bounds__(256) void cvt_f32_to_f16(const float* __restrict__ in,
                                                      _Float16* __restrict__ out,
                                                      int n4) {
    int i = (blockIdx.x * 256 + threadIdx.x);
    if (i < n4) {
        float4 v = ((const float4*)in)[i];
        _Float16* p = out + i * 4;
        p[0] = (_Float16)v.x; p[1] = (_Float16)v.y;
        p[2] = (_Float16)v.z; p[3] = (_Float16)v.w;
    }
}

// ---------------------------------------------------------------------------
// zero scratch state (16 bytes / thread)
// ---------------------------------------------------------------------------
__global__ __launch_bounds__(256) void zero_ws(float4* __restrict__ p, int n16) {
    int i = blockIdx.x * 256 + threadIdx.x;
    if (i < n16) p[i] = make_float4(0.f, 0.f, 0.f, 0.f);
}

// ---------------------------------------------------------------------------
// Fused LSTM step: gate GEMM (WMMA, two K regions) + peephole cell epilogue.
// Wave covers rows [r0, r0+32) x hidden cols [hc0, hc0+16) x 4 gates.
// ---------------------------------------------------------------------------
__global__ __launch_bounds__(256)
void lstm_step(const _Float16* __restrict__ A1, int aoff1, int astr1, int k1,
               const _Float16* __restrict__ Wx,
               const _Float16* __restrict__ A2,
               const _Float16* __restrict__ Wh,
               float* __restrict__ c,
               const float* __restrict__ Wc,
               const float* __restrict__ bx,
               _Float16* __restrict__ hOut,
               _Float16* __restrict__ hStore, int soff, int sstr) {
    const int lane = threadIdx.x & 31;
    const int w    = threadIdx.x >> 5;            // 0..7 -> hidden-col tile
    const int r0   = blockIdx.x * 32;             // rows r0..r0+31 (whole block)
    const int hc0  = blockIdx.y * 128 + w * 16;   // this wave's 16 hidden cols

    // A fragment: lane holds row (lane&15); halves j = K kb+j.
    const int arow = (lane & 15);
    const int kb   = (lane >> 4) * 16;
    // B fragment: lane holds column (lane&15); halves j = K ks+j.
    const int ks   = (lane >> 4) * 16;
    const int ncol = lane & 15;

    v8f acc0[4] = {v8f{}, v8f{}, v8f{}, v8f{}};   // rows r0..r0+15 : i,f,g,o
    v8f acc1[4] = {v8f{}, v8f{}, v8f{}, v8f{}};   // rows r0+16..r0+31

    // ---- K region 1: input projection ----
    {
        const _Float16* aRow0 = A1 + (size_t)(r0 + arow) * astr1 + aoff1;
        const _Float16* aRow1 = A1 + (size_t)(r0 + 16 + arow) * astr1 + aoff1;
        for (int kk = 0; kk < k1; kk += 32) {
            v16h Af0 = load_afrag(aRow0 + kk + kb);
            v16h Af1 = load_afrag(aRow1 + kk + kb);
#pragma unroll
            for (int g = 0; g < 4; ++g) {
                const _Float16* colp = Wx + (size_t)(g * Hsz + hc0 + ncol) * k1 + ks;
                v16h Bf = *(const v16h*)(colp + kk);
                __builtin_prefetch(colp + kk + 64, 0, 2);  // speculative, branch-free
                acc0[g] = __builtin_amdgcn_wmma_f32_16x16x32_f16(
                    false, Af0, false, Bf, (short)0, acc0[g], false, false);
                acc1[g] = __builtin_amdgcn_wmma_f32_16x16x32_f16(
                    false, Af1, false, Bf, (short)0, acc1[g], false, false);
            }
        }
    }
    // ---- K region 2: recurrent projection (K = 512) ----
    {
        const _Float16* aRow0 = A2 + (size_t)(r0 + arow) * Hsz;
        const _Float16* aRow1 = A2 + (size_t)(r0 + 16 + arow) * Hsz;
        for (int kk = 0; kk < Hsz; kk += 32) {
            v16h Af0 = load_afrag(aRow0 + kk + kb);
            v16h Af1 = load_afrag(aRow1 + kk + kb);
#pragma unroll
            for (int g = 0; g < 4; ++g) {
                const _Float16* colp = Wh + (size_t)(g * Hsz + hc0 + ncol) * Hsz + ks;
                v16h Bf = *(const v16h*)(colp + kk);
                __builtin_prefetch(colp + kk + 64, 0, 2);  // speculative, branch-free
                acc0[g] = __builtin_amdgcn_wmma_f32_16x16x32_f16(
                    false, Af0, false, Bf, (short)0, acc0[g], false, false);
                acc1[g] = __builtin_amdgcn_wmma_f32_16x16x32_f16(
                    false, Af1, false, Bf, (short)0, acc1[g], false, false);
            }
        }
    }

    // ---- Peephole cell epilogue in accumulator registers ----
    // C/D layout: lanes 0-15: n=lane, m=vgpr r; lanes 16-31: n=lane-16, m=r+8.
    const int col = hc0 + ncol;
    const float bi = bx[col];
    const float bf = bx[Hsz + col];
    const float bg = bx[2 * Hsz + col];
    const float bo = bx[3 * Hsz + col];
    const float wc0 = Wc[col];
    const float wc1 = Wc[Hsz + col];
    const float wc2 = Wc[2 * Hsz + col];
#pragma unroll
    for (int mt = 0; mt < 2; ++mt) {
        const v8f* acc = mt ? acc1 : acc0;
#pragma unroll
        for (int r = 0; r < 8; ++r) {
            const int m   = (lane >> 4) * 8 + r;
            const int row = r0 + mt * 16 + m;
            const size_t idx = (size_t)row * Hsz + col;
            const float cp = c[idx];
            const float iv = sigmoid_f(acc[0][r] + bi + cp * wc0);
            const float fv = sigmoid_f(acc[1][r] + bf + cp * wc1);
            const float gv = tanh_f(acc[2][r] + bg);
            const float cn = fv * cp + iv * gv;
            const float ov = sigmoid_f(acc[3][r] + bo + cn * wc2);
            const float hv = ov * tanh_f(cn);
            c[idx]    = cn;
            hOut[idx] = (_Float16)hv;
            if (hStore) hStore[(size_t)row * sstr + soff + col] = (_Float16)hv;
        }
    }
}

// ---------------------------------------------------------------------------
// Final FC: out[b,k] = h1[b,:] . Wfc[k,:] + bfc[k]   (256 x 8, K = 512)
// ---------------------------------------------------------------------------
__global__ __launch_bounds__(256)
void fc_kernel(const _Float16* __restrict__ h, const float* __restrict__ Wfc,
               const float* __restrict__ bfc, float* __restrict__ out) {
    int idx = blockIdx.x * 256 + threadIdx.x;   // 0..2047
    int b = idx >> 3, k = idx & 7;
    const _Float16* hr = h + (size_t)b * Hsz;
    const float* wr = Wfc + (size_t)k * Hsz;
    float s = 0.f;
#pragma unroll 8
    for (int j = 0; j < Hsz; ++j) s += (float)hr[j] * wr[j];
    out[idx] = s + bfc[k];
}

// ---------------------------------------------------------------------------
// Launcher
// ---------------------------------------------------------------------------
extern "C" void kernel_launch(void* const* d_in, const int* in_sizes, int n_in,
                              void* d_out, int out_size, void* d_ws, size_t ws_size,
                              hipStream_t stream) {
    const float* x   = (const float*)d_in[0];
    const float* Wx0 = (const float*)d_in[1];
    const float* bx0 = (const float*)d_in[2];
    const float* Wh0 = (const float*)d_in[3];
    const float* Wc0 = (const float*)d_in[4];
    const float* Wx1 = (const float*)d_in[5];
    const float* bx1 = (const float*)d_in[6];
    const float* Wh1 = (const float*)d_in[7];
    const float* Wc1 = (const float*)d_in[8];
    const float* Wfc = (const float*)d_in[9];
    const float* bfc = (const float*)d_in[10];
    float* out = (float*)d_out;

    char* ws = (char*)d_ws;
    // workspace layout (bytes)
    _Float16* Xh    = (_Float16*)(ws);                       // 256*128*256 h = 16 MiB
    _Float16* Wx0h  = (_Float16*)(ws + 16777216);            // 2048*256 h
    _Float16* Wh0h  = (_Float16*)(ws + 17825792);            // 2048*512 h
    _Float16* Wx1h  = (_Float16*)(ws + 19922944);            // 2048*512 h
    _Float16* Wh1h  = (_Float16*)(ws + 22020096);            // 2048*512 h
    _Float16* H0all = (_Float16*)(ws + 24117248);            // 256*128*512 h = 32 MiB
    _Float16* h0a   = (_Float16*)(ws + 57671680);            // 256*512 h
    _Float16* h0b   = (_Float16*)(ws + 57933824);
    _Float16* h1a   = (_Float16*)(ws + 58195968);
    _Float16* h1b   = (_Float16*)(ws + 58458112);
    float*    c0    = (float*)   (ws + 58720256);            // 256*512 f32
    float*    c1    = (float*)   (ws + 59244544);
    // total = 59,768,832 bytes

    // zero state region (h0a .. end of c1): 2,097,152 bytes = 131072 float4
    zero_ws<<<512, 256, 0, stream>>>((float4*)h0a, 131072);

    // f16 conversions
    cvt_f32_to_f16<<<8192, 256, 0, stream>>>(x,   Xh,   Bsz * Tsz * INsz / 4);
    cvt_f32_to_f16<<<512,  256, 0, stream>>>(Wx0, Wx0h, 2048 * 256 / 4);
    cvt_f32_to_f16<<<1024, 256, 0, stream>>>(Wh0, Wh0h, 2048 * 512 / 4);
    cvt_f32_to_f16<<<1024, 256, 0, stream>>>(Wx1, Wx1h, 2048 * 512 / 4);
    cvt_f32_to_f16<<<1024, 256, 0, stream>>>(Wh1, Wh1h, 2048 * 512 / 4);

    dim3 grid(Bsz / 32, Hsz / 128);  // (8, 4)

    // layer 0: A1 = x_t (row b at Xh + b*T*IN + t*IN), K1 = 256
    for (int t = 0; t < Tsz; ++t) {
        const _Float16* hin  = (t & 1) ? h0b : h0a;
        _Float16*       hout = (t & 1) ? h0a : h0b;
        lstm_step<<<grid, 256, 0, stream>>>(
            Xh, t * INsz, Tsz * INsz, INsz, Wx0h,
            hin, Wh0h, c0, Wc0, bx0,
            hout, H0all, t * Hsz, Tsz * Hsz);
    }
    // layer 1: A1 = h0_t history (row b at H0all + b*T*H + t*H), K1 = 512
    for (int t = 0; t < Tsz; ++t) {
        const _Float16* hin  = (t & 1) ? h1b : h1a;
        _Float16*       hout = (t & 1) ? h1a : h1b;
        lstm_step<<<grid, 256, 0, stream>>>(
            H0all, t * Hsz, Tsz * Hsz, Hsz, Wx1h,
            hin, Wh1h, c1, Wc1, bx1,
            hout, (_Float16*)nullptr, 0, 0);
    }

    // final h1 after t=127 (odd) was written to h1a
    fc_kernel<<<8, 256, 0, stream>>>(h1a, Wfc, bfc, out);
}
